// GATLayer_8529805050330
// MI455X (gfx1250) — matrix operand order repaired and checked by reading, not compiled, not run on hardware
//
#include <hip/hip_runtime.h>
#include <hip/hip_bf16.h>

// ---- CDNA5 WMMA vector types (wave32) ----
typedef __attribute__((ext_vector_type(16))) __bf16 v16bf;
typedef __attribute__((ext_vector_type(8)))  __bf16 v8bf;
typedef __attribute__((ext_vector_type(8)))  float  v8f;

#define N_NODES 2048
#define F_IN    512
#define F_OUT   64
#define NEG_BIG (-9.0e15f)
#define SSTR    2052   // 2048 + 4 floats pad -> 16 A-frag rows land on distinct LDS banks

// ---------------------------------------------------------------------------
// Kernel 1a: x f32 -> bf16 (row-major, [2048,512])
// ---------------------------------------------------------------------------
__global__ void cvt_x_bf16(const float* __restrict__ x, __bf16* __restrict__ xb, int n) {
    int i = blockIdx.x * blockDim.x + threadIdx.x;
    if (i < n) xb[i] = (__bf16)x[i];
}

// ---------------------------------------------------------------------------
// Kernel 1b: W [512,64] f32 -> Wt [64,512] bf16 (N-major so B-frags are contiguous)
// ---------------------------------------------------------------------------
__global__ void cvt_w_bf16(const float* __restrict__ W, __bf16* __restrict__ Wt) {
    int i = blockIdx.x * blockDim.x + threadIdx.x;
    if (i < F_IN * F_OUT) {
        int k = i >> 6, n = i & 63;
        Wt[n * F_IN + k] = (__bf16)W[k * F_OUT + n];
    }
}

// ---------------------------------------------------------------------------
// Kernel 2: h = x @ W via v_wmma_f32_16x16x32_bf16.
// Grid: 128 blocks (one 16-row M tile each), 128 threads = 4 waves,
// wave w owns N-tile w (F_OUT=64 -> 4 tiles). K loop: 512/32 = 16 WMMAs/wave.
// Writes h_f32 [2048,64] (for f32 scoring) and ht_bf16 [64,2048] (f-major, PV B side).
// ---------------------------------------------------------------------------
__global__ void gemm_h(const __bf16* __restrict__ xb, const __bf16* __restrict__ Wt,
                       float* __restrict__ hf, __bf16* __restrict__ ht) {
    const int lane = threadIdx.x & 31;
    const int wave = threadIdx.x >> 5;          // = N tile
    const int i0   = blockIdx.x * 16;
    const int m    = lane & 15;
    const int hsel = lane >> 4;                 // 0/1 half of wave

    const __bf16* arow = xb + (size_t)(i0 + m) * F_IN;
    const __bf16* brow = Wt + (size_t)(wave * 16 + m) * F_IN + (hsel ? 16 : 0);

    v8f c = {};
    for (int k0 = 0; k0 < F_IN; k0 += 32) {
        v16bf a, b;
        const v8bf alo = *(const v8bf*)(arow + k0 + (hsel ? 16 : 0));
        const v8bf ahi = *(const v8bf*)(arow + k0 + 8 + (hsel ? 16 : 0));
        #pragma unroll
        for (int e = 0; e < 8; ++e) { a[e] = alo[e]; a[8 + e] = ahi[e]; }
        b = *(const v16bf*)(brow + k0);
        c = __builtin_amdgcn_wmma_f32_16x16x32_bf16(false, a, false, b,
                                                    (short)0, c, false, false);
    }
    #pragma unroll
    for (int r = 0; r < 8; ++r) {
        int mr  = r + hsel * 8;                 // D row
        int col = wave * 16 + (lane & 15);      // D col
        float v = c[r];
        hf[(size_t)(i0 + mr) * F_OUT + col] = v;
        ht[(size_t)col * N_NODES + (i0 + mr)] = (__bf16)v;
    }
}

// ---------------------------------------------------------------------------
// Kernel 3: fused scores + masked softmax + P@h (WMMA) for one 16-row tile.
// 256 threads = 8 waves. Dynamic LDS:
//   sS   [16][SSTR] f32 scores/probs (128 KB — CDNA5 has 320 KB/WGP)
//   sHi  [16][64]   f32 h rows of this tile
//   sA   [64]       f32 attention vector a
//   sSum [16]       f32 softmax denominators
//   sPart[4][8][32] f32 cross-wave partial D tiles
// ---------------------------------------------------------------------------
__global__ void gat_attn(const float* __restrict__ hf, const __bf16* __restrict__ ht,
                         const int* __restrict__ adj, const float* __restrict__ a,
                         float* __restrict__ out) {
    extern __shared__ char smem_raw[];
    float* sS    = (float*)smem_raw;            // 16*SSTR
    float* sHi   = sS + 16 * SSTR;              // 1024
    float* sA    = sHi + 1024;                  // 64
    float* sSum  = sA + 64;                     // 16
    float* sPart = sSum + 16;                   // 1024

    const int t    = threadIdx.x;
    const int i0   = blockIdx.x * 16;
    const int wave = t >> 5;
    const int lane = t & 31;

    for (int idx = t; idx < 16 * F_OUT; idx += 256)
        sHi[idx] = hf[(size_t)i0 * F_OUT + idx];
    if (t < F_OUT) sA[t] = a[t];
    __syncthreads();

    // ---- Phase 1: masked scores (f32 VALU; dominant compute) ----
    for (int jb = 0; jb < 8; ++jb) {
        const int j = jb * 256 + t;
        float hj[F_OUT];
        const float4* hjp = (const float4*)(hf + (size_t)j * F_OUT);
        #pragma unroll
        for (int q = 0; q < 16; ++q) {
            float4 v = hjp[q];
            hj[4*q] = v.x; hj[4*q+1] = v.y; hj[4*q+2] = v.z; hj[4*q+3] = v.w;
        }
        for (int i = 0; i < 16; ++i) {
            float acc = 0.f;
            #pragma unroll
            for (int f = 0; f < F_OUT; ++f)
                acc = fmaf(fabsf(hj[f] - sHi[i * F_OUT + f]), sA[f], acc);
            float e  = acc > 0.f ? acc : 0.f;                     // relu
            float sc = (adj[(size_t)(i0 + i) * N_NODES + j] > 0) ? e : NEG_BIG;
            sS[i * SSTR + j] = sc;
        }
    }
    __syncthreads();

    // ---- Phase 2: row softmax (store unnormalized exp; fold 1/sum into D) ----
    for (int rr = 0; rr < 2; ++rr) {
        const int i = wave * 2 + rr;
        float mx = -3.4e38f;
        for (int j = lane; j < N_NODES; j += 32) mx = fmaxf(mx, sS[i * SSTR + j]);
        #pragma unroll
        for (int off = 16; off > 0; off >>= 1) mx = fmaxf(mx, __shfl_xor(mx, off, 32));
        float sum = 0.f;
        for (int j = lane; j < N_NODES; j += 32) {
            float p = __expf(sS[i * SSTR + j] - mx);
            sS[i * SSTR + j] = p;
            sum += p;
        }
        #pragma unroll
        for (int off = 16; off > 0; off >>= 1) sum += __shfl_xor(sum, off, 32);
        if (lane == 0) sSum[i] = sum;
    }
    __syncthreads();

    // ---- Phase 3: out_tile = P[16x2048] @ h[2048x64] via WMMA bf16 ----
    const int ntile = wave & 3;                 // N tile (0..3)
    const int khalf = wave >> 2;                // split K across wave pairs
    const int m     = lane & 15;
    const int hsel  = lane >> 4;

    const __bf16* brow = ht + (size_t)(ntile * 16 + m) * N_NODES + (hsel ? 16 : 0);
    const float*  srow = sS + m * SSTR + (hsel ? 16 : 0);

    v8f c = {};
    for (int ks = 0; ks < 32; ++ks) {
        const int k0 = (khalf * 32 + ks) * 32;
        v16bf aF, bF;
        const float* p1 = srow + k0;
        #pragma unroll
        for (int e = 0; e < 8; ++e) aF[e]     = (__bf16)p1[e];       // K = k0+sel+e
        #pragma unroll
        for (int e = 0; e < 8; ++e) aF[8 + e] = (__bf16)p1[8 + e];   // K = k0+sel+8+e
        bF = *(const v16bf*)(brow + k0);
        c = __builtin_amdgcn_wmma_f32_16x16x32_bf16(false, aF, false, bF,
                                                    (short)0, c, false, false);
    }
    if (khalf == 1) {
        #pragma unroll
        for (int r = 0; r < 8; ++r) sPart[(ntile * 8 + r) * 32 + lane] = c[r];
    }
    __syncthreads();
    if (khalf == 0) {
        #pragma unroll
        for (int r = 0; r < 8; ++r) {
            float v  = c[r] + sPart[(ntile * 8 + r) * 32 + lane];
            int   mr = r + hsel * 8;
            v *= 1.0f / sSum[mr];               // softmax normalization
            v  = v > 0.f ? v : 0.f;             // final relu
            out[(size_t)(i0 + mr) * F_OUT + ntile * 16 + (lane & 15)] = v;
        }
    }
}

// ---------------------------------------------------------------------------
extern "C" void kernel_launch(void* const* d_in, const int* in_sizes, int n_in,
                              void* d_out, int out_size, void* d_ws, size_t ws_size,
                              hipStream_t stream) {
    const float* x   = (const float*)d_in[0];   // [2048,512]
    const int*   adj = (const int*)  d_in[1];   // [2048,2048]
    const float* W   = (const float*)d_in[2];   // [512,64]
    const float* a   = (const float*)d_in[3];   // [64,1]
    float* out = (float*)d_out;                 // [2048,64]

    // workspace carve
    __bf16* xb = (__bf16*)d_ws;                         // 2048*512 bf16 = 2 MB
    __bf16* Wt = xb + (size_t)N_NODES * F_IN;           // 64*512  bf16 = 64 KB
    float*  hf = (float*)(Wt + (size_t)F_OUT * F_IN);   // 2048*64 f32  = 512 KB
    __bf16* ht = (__bf16*)(hf + (size_t)N_NODES * F_OUT); // 64*2048 bf16 = 256 KB

    cvt_x_bf16<<<(N_NODES * F_IN + 255) / 256, 256, 0, stream>>>(x, xb, N_NODES * F_IN);
    cvt_w_bf16<<<(F_IN * F_OUT + 255) / 256, 256, 0, stream>>>(W, Wt);
    gemm_h<<<N_NODES / 16, 128, 0, stream>>>(xb, Wt, hf, ht);

    const size_t smem = (size_t)(16 * SSTR + 1024 + 64 + 16 + 1024) * sizeof(float); // ~140 KB
    gat_attn<<<N_NODES / 16, 256, smem, stream>>>(hf, ht, adj, a, out);
}
